// CTRGC_36129264894592
// MI455X (gfx1250) — matrix operand order repaired and checked by reading, not compiled, not run on hardware
//
#include <hip/hip_runtime.h>
#include <hip/hip_bf16.h>
#include <math.h>

#define N_   64
#define C_   64
#define T_   256
#define V_   25
#define VP   32            // padded V
#define REL_ 8
#define O_   64
#define TT   32            // t-tile per block
#define TBLK (T_ / TT)     // 8 t-tiles
#define P_   (TT * VP)     // 1024 staged positions per block

#define CPITCH 72          // xst row pitch (halves): 144B, 16B-aligned, bank-staggered
#define X3P    1032        // x3s row pitch (halves): 2064B, 16B-aligned
#define DP     40          // d tile row pitch (halves): 80B, 16B-aligned
#define DTILE  (VP * DP)   // halves per (n,o) adjacency tile (1280 -> 2560B)

typedef __attribute__((ext_vector_type(16))) __bf16 v16bf;
typedef __attribute__((ext_vector_type(8)))  __bf16 v8bf;
typedef __attribute__((ext_vector_type(4)))  __bf16 v4bf;
typedef __attribute__((ext_vector_type(8)))  float  v8f;
typedef __attribute__((ext_vector_type(4)))  unsigned int u32x4;
typedef __attribute__((ext_vector_type(8)))  int i32x8;
typedef __attribute__((ext_vector_type(4)))  int i32x4;

union Frag16 { v16bf v; v8bf h[2]; };

// ---------------------------------------------------------------------------
// TDM: 1-D async copy global -> LDS, 8-byte elements.  D# per ISA 8.3/8.4.
// ---------------------------------------------------------------------------
#if defined(__gfx1250__) && __has_builtin(__builtin_amdgcn_tensor_load_to_lds)
#define HAVE_TDM 1
__device__ __forceinline__ void tdm_load_1d(unsigned lds_off, const void* gptr,
                                            unsigned nbytes) {
  const unsigned long long ga = (unsigned long long)(uintptr_t)gptr;
  const unsigned elems = nbytes >> 3;                 // 8B units
  u32x4 g0;
  g0[0] = 1u;                                         // count=1 valid, user D#
  g0[1] = lds_off;                                    // LDS byte address
  g0[2] = (unsigned)(ga & 0xffffffffu);               // global_addr[31:0]
  g0[3] = (unsigned)((ga >> 32) & 0x01ffffffu) | (2u << 30);  // [56:32] | type=2
  i32x8 g1;
  g1[0] = (int)(3u << 16);                            // wg_mask=0, data_size=8B
  g1[1] = (int)((elems & 0xffffu) << 16);             // tensor_dim0[15:0]
  g1[2] = (int)(((elems >> 16) & 0xffffu) | (1u << 16)); // dim0[31:16]|dim1=1
  g1[3] = (int)((elems & 0xffffu) << 16);             // dim1 hi=0 | tile_dim0
  g1[4] = 0;                                          // tile_dim1=0, tile_dim2=0
  g1[5] = (int)elems;                                 // dim0_stride[31:0]
  g1[6] = 0;                                          // stride hi | dim1_stride lo
  g1[7] = 0;
  i32x4 z4 = {0, 0, 0, 0};
#if __clang_major__ >= 23
  i32x8 z8 = {0, 0, 0, 0, 0, 0, 0, 0};
  __builtin_amdgcn_tensor_load_to_lds(g0, g1, z4, z4, z8, 0);
#else
  __builtin_amdgcn_tensor_load_to_lds(g0, g1, z4, z4, 0);
#endif
}
__device__ __forceinline__ void tdm_wait() {
#if __has_builtin(__builtin_amdgcn_s_wait_tensorcnt)
  __builtin_amdgcn_s_wait_tensorcnt(0);
#else
  asm volatile("s_wait_tensorcnt 0" ::: "memory");
#endif
}
#else
#define HAVE_TDM 0
#endif

// ---------------------------------------------------------------------------
// Kernel 1: xm[n,c,v] = mean_t x[n,c,t,v].  One wave32 per (n,c).
// ---------------------------------------------------------------------------
__global__ void ctrgc_mean(const float* __restrict__ x, float* __restrict__ xm) {
  const int nc   = blockIdx.x;
  const int lane = threadIdx.x;      // 32 threads
  float acc[V_];
#pragma unroll
  for (int v = 0; v < V_; ++v) acc[v] = 0.f;
  const float* base = x + (size_t)nc * T_ * V_;
  for (int t = lane; t < T_; t += 32) {
    const float* row = base + t * V_;
#pragma unroll
    for (int v = 0; v < V_; ++v) acc[v] += row[v];
  }
#pragma unroll
  for (int off = 16; off > 0; off >>= 1) {
#pragma unroll
    for (int v = 0; v < V_; ++v) acc[v] += __shfl_down(acc[v], off, 32);
  }
  if (lane == 0) {
#pragma unroll
    for (int v = 0; v < V_; ++v) xm[nc * V_ + v] = acc[v] * (1.f / T_);
  }
}

// ---------------------------------------------------------------------------
// Kernel 2: dynamic topology, emitted as bf16 tiles already transposed/padded
// into the exact LDS layout consumed by kernel 3:
//   dbf[(n*O+o)*DTILE + v*DP + u] = (u<25 && v<25) ? A[u,v]+b4[o]+sum_r ... : 0
// ---------------------------------------------------------------------------
__global__ void ctrgc_topo(const float* __restrict__ xm, const float* __restrict__ A,
                           const float* __restrict__ w1, const float* __restrict__ b1,
                           const float* __restrict__ w2, const float* __restrict__ b2,
                           const float* __restrict__ w4, const float* __restrict__ b4,
                           __bf16* __restrict__ dbf) {
  const int n   = blockIdx.x;
  const int tid = threadIdx.x;
  __shared__ float xms[C_ * V_];
  __shared__ float x1s[REL_ * V_];
  __shared__ float x2s[REL_ * V_];
  __shared__ float adjs[REL_ * V_ * V_];

  for (int i = tid; i < C_ * V_; i += 256) xms[i] = xm[(size_t)n * C_ * V_ + i];
  __syncthreads();

  for (int i = tid; i < REL_ * V_; i += 256) {
    const int r = i / V_, v = i % V_;
    float s1 = b1[r], s2 = b2[r];
#pragma unroll 8
    for (int c = 0; c < C_; ++c) {
      const float xv = xms[c * V_ + v];
      s1 += xv * w1[r * C_ + c];
      s2 += xv * w2[r * C_ + c];
    }
    x1s[i] = s1; x2s[i] = s2;
  }
  __syncthreads();

  for (int i = tid; i < REL_ * V_ * V_; i += 256) {
    const int r = i / (V_ * V_), rem = i % (V_ * V_);
    const int u = rem / V_, v = rem % V_;
    adjs[i] = tanhf(x1s[r * V_ + u] - x2s[r * V_ + v]);
  }
  __syncthreads();

  for (int i = tid; i < O_ * DTILE; i += 256) {
    const int o = i / DTILE, rem = i % DTILE;
    const int vv = rem / DP, u = rem % DP;
    float val = 0.f;
    if (u < V_ && vv < V_) {
      float s = b4[o];
#pragma unroll
      for (int r = 0; r < REL_; ++r)
        s += w4[o * REL_ + r] * adjs[r * V_ * V_ + u * V_ + vv];
      val = s + A[u * V_ + vv];          // ALPHA == 1
    }
    dbf[(size_t)n * O_ * DTILE + i] = (__bf16)val;
  }
}

// ---------------------------------------------------------------------------
// Kernel 3: fused  x3 = w3 @ x + b3 ; out[n,o,t,v] = sum_u x3[n,o,t,u]*d[n,o,u,v]
// Block = (n, 32-t tile), 8 wave32.  WMMA bf16 for both GEMMs; adjacency tiles
// streamed into double-buffered LDS by the Tensor Data Mover (TENSORcnt).
// ---------------------------------------------------------------------------
__global__ __launch_bounds__(256) void ctrgc_main(
    const float* __restrict__ x, const float* __restrict__ w3,
    const float* __restrict__ b3, const __bf16* __restrict__ dbf,
    float* __restrict__ out) {
  const int n    = blockIdx.x / TBLK;
  const int t0   = (blockIdx.x % TBLK) * TT;
  const int tid  = threadIdx.x;
  const int wave = tid >> 5;
  const int lane = tid & 31;
  const int g    = lane >> 4;    // half-wave operand group
  const int ln   = lane & 15;

  __shared__ __align__(16) __bf16 xst[P_][CPITCH];    // x tile (pos,chan)  144 KB
  __shared__ __align__(16) __bf16 w3s[O_ * C_];       // w3 row-major         8 KB
  __shared__ __align__(16) __bf16 x3s[16][X3P];       // x3, one 16-o tile 32.25 KB
  __shared__ __align__(16) __bf16 dlst[2][16][DTILE]; // adjacency, dbl-buf  80 KB
  __shared__ float b3s[O_];

  const __bf16* dsrc = dbf + (size_t)n * O_ * DTILE;  // this n's 4 o-tiles

#if HAVE_TDM
  // kick off o-tile 0 DMA before staging x, so it overlaps the heavy loads
  if (wave == 0)
    tdm_load_1d((unsigned)(uintptr_t)&dlst[0][0][0], dsrc, 16 * DTILE * 2);
#endif

  // ---- stage x[n,:,t-tile,:] transposed to (position, channel), bf16 ----
  for (int i = tid; i < P_ * (C_ / 4); i += 256) {
    const int p  = i & (P_ - 1);
    const int cg = i >> 10;            // P_ == 1024
    const int tl = p >> 5, v = p & (VP - 1);
    v4bf pk;
    if (v < V_) {
      const float* gp = x + ((size_t)(n * C_ + cg * 4) * T_ + (t0 + tl)) * V_ + v;
      const size_t cs = (size_t)T_ * V_;
      pk[0] = (__bf16)gp[0];
      pk[1] = (__bf16)gp[cs];
      pk[2] = (__bf16)gp[2 * cs];
      pk[3] = (__bf16)gp[3 * cs];
    } else {
      pk[0] = pk[1] = pk[2] = pk[3] = (__bf16)0.f;
    }
    *(v4bf*)&xst[p][cg * 4] = pk;
  }
  for (int i = tid; i < (O_ * C_) / 4; i += 256) {
    const float* gp = w3 + i * 4;
    v4bf pk;
    pk[0] = (__bf16)gp[0]; pk[1] = (__bf16)gp[1];
    pk[2] = (__bf16)gp[2]; pk[3] = (__bf16)gp[3];
    *(v4bf*)&w3s[i * 4] = pk;
  }
  if (tid < O_) b3s[tid] = b3[tid];
#if !HAVE_TDM
  // fallback: synchronous vector copy of o-tile 0
  {
    const v8bf* s = (const v8bf*)dsrc;
    v8bf* d = (v8bf*)&dlst[0][0][0];
    for (int i = tid; i < 16 * DTILE / 8; i += 256) d[i] = s[i];
  }
#endif
  __syncthreads();

  for (int ot = 0; ot < O_ / 16; ++ot) {
    const int o0  = ot * 16;
    const int buf = ot & 1;

#if HAVE_TDM
    if (wave == 0) tdm_wait();         // current buffer's DMA complete
    __syncthreads();
    if (wave == 0 && ot < O_ / 16 - 1) // overlap next tile's DMA with compute
      tdm_load_1d((unsigned)(uintptr_t)&dlst[buf ^ 1][0][0],
                  dsrc + (size_t)(o0 + 16) * DTILE, 16 * DTILE * 2);
#else
    if (ot > 0) {
      const v8bf* s = (const v8bf*)(dsrc + (size_t)o0 * DTILE);
      v8bf* d = (v8bf*)&dlst[buf][0][0];
      for (int i = tid; i < 16 * DTILE / 8; i += 256) d[i] = s[i];
      __syncthreads();
    }
#endif

    // ---- GEMM A: x3 tile = w3[o-tile,:] (16x64) @ x (64 x P_) ----
    Frag16 afr[2];
    const __bf16* wrow = &w3s[(o0 + ln) * C_];
#pragma unroll
    for (int ks = 0; ks < 2; ++ks) {
      afr[ks].h[0] = *(const v8bf*)(wrow + 32 * ks + 8 * g);      // k=8g..8g+7
      afr[ks].h[1] = *(const v8bf*)(wrow + 32 * ks + 16 + 8 * g); // k=16+8g..+7
    }
    for (int pt = wave; pt < P_ / 16; pt += 8) {
      const __bf16* xrow = &xst[pt * 16 + ln][0];
      v8f acc;
#pragma unroll
      for (int r = 0; r < 8; ++r) acc[r] = 0.f;
#pragma unroll
      for (int ks = 0; ks < 2; ++ks) {
        Frag16 bfr;                                                // k=8g..+7, 16+8g..+7
        bfr.h[0] = *(const v8bf*)(xrow + 32 * ks + 8 * g);
        bfr.h[1] = *(const v8bf*)(xrow + 32 * ks + 16 + 8 * g);
        acc = __builtin_amdgcn_wmma_f32_16x16x32_bf16(false, afr[ks].v, false, bfr.v,
                                                      (short)0, acc, false, false);
      }
#pragma unroll
      for (int r = 0; r < 8; ++r) {                 // D: (m = r + 8g, col = ln)
        const int m = r + 8 * g;
        x3s[m][pt * 16 + ln] = (__bf16)(acc[r] + b3s[o0 + m]);
      }
    }
    __syncthreads();

    // ---- GEMM B: per o, out (TT x VP) = x3[o] (TT x VP) @ d[o] (VP x VP) ----
    for (int q = wave; q < 16 * 2 * 2; q += 8) {
      const int ol = q >> 2, mt = (q >> 1) & 1, nt = q & 1;
      Frag16 afr2, bfr2;
      const __bf16* xr = &x3s[ol][(mt * 16 + ln) * VP];
      afr2.h[0] = *(const v8bf*)(xr + 8 * g);
      afr2.h[1] = *(const v8bf*)(xr + 16 + 8 * g);
      const __bf16* dr = &dlst[buf][ol][(nt * 16 + ln) * DP];
      bfr2.h[0] = *(const v8bf*)(dr + 8 * g);
      bfr2.h[1] = *(const v8bf*)(dr + 16 + 8 * g);
      v8f acc;
#pragma unroll
      for (int r = 0; r < 8; ++r) acc[r] = 0.f;
      acc = __builtin_amdgcn_wmma_f32_16x16x32_bf16(false, afr2.v, false, bfr2.v,
                                                    (short)0, acc, false, false);
      const int vv = nt * 16 + ln;
      if (vv < V_) {
        const int o = o0 + ol;
#pragma unroll
        for (int r = 0; r < 8; ++r) {
          const int t = t0 + mt * 16 + r + 8 * g;
          out[((size_t)(n * O_ + o) * T_ + t) * V_ + vv] = acc[r];
        }
      }
    }
    __syncthreads();
  }
}

// ---------------------------------------------------------------------------
extern "C" void kernel_launch(void* const* d_in, const int* in_sizes, int n_in,
                              void* d_out, int out_size, void* d_ws, size_t ws_size,
                              hipStream_t stream) {
  (void)in_sizes; (void)n_in; (void)out_size; (void)ws_size;
  const float* x  = (const float*)d_in[0];
  const float* A  = (const float*)d_in[1];
  const float* w1 = (const float*)d_in[2];
  const float* b1 = (const float*)d_in[3];
  const float* w2 = (const float*)d_in[4];
  const float* b2 = (const float*)d_in[5];
  const float* w3 = (const float*)d_in[6];
  const float* b3 = (const float*)d_in[7];
  const float* w4 = (const float*)d_in[8];
  const float* b4 = (const float*)d_in[9];
  float* out = (float*)d_out;
  float*  xm  = (float*)d_ws;                               // N*C*V f32 (400 KB)
  __bf16* dbf = (__bf16*)((char*)d_ws + (size_t)N_ * C_ * V_ * sizeof(float));
                                                            // N*O*DTILE bf16 (10.5 MB)
  ctrgc_mean<<<N_ * C_, 32, 0, stream>>>(x, xm);
  ctrgc_topo<<<N_, 256, 0, stream>>>(xm, A, w1, b1, w2, b2, w4, b4, dbf);
  ctrgc_main<<<N_ * TBLK, 256, 0, stream>>>(x, w3, b3, dbf, out);
}
